// SuperpixelConv_old_68633577390528
// MI455X (gfx1250) — matrix operand order repaired and checked by reading, not compiled, not run on hardware
//
#include <hip/hip_runtime.h>
#include <hip/hip_bf16.h>

// ---------------------------------------------------------------------------
// SuperpixelConv for MI455X (gfx1250, wave32, WMMA).
//
// One wave (= one 32-thread workgroup) owns a 16-pixel row tile (h, w0..w0+15).
//   Pass 1: scores[q, key] via v_wmma_f32_16x16x32_f16, K = channels (8 chunks
//           of 32), keys = two 16-column tiles at clip(w0-2) / clip(w0+2)
//           covering the 20-column union of all clamped 5-windows, x 5 rows.
//   Middle: softmax + sims reweight + L1 norm per query (lanes 0..15).
//   Pass 2: out[q, c] via WMMA with a sparse 16x128 kernel-band A matrix
//           (25 nonzeros per row) against the staged 100-key neighborhood.
// LDS stages x as channel-pair-interleaved f16 so WMMA A/B fragments load as
// ds_load_b32 per dword, matching the ISA 16-bit A(16x32)/B(32x16) layouts.
// ---------------------------------------------------------------------------

typedef __attribute__((ext_vector_type(16))) _Float16 v16h;
typedef __attribute__((ext_vector_type(8)))  float    v8f;

constexpr int C    = 256;
constexpr int H    = 128;
constexpr int W    = 128;
constexpr int HW   = H * W;
constexpr int WSZ  = 5;              // window size
constexpr int NK   = WSZ * WSZ;      // 25 neighbors
constexpr int TILE = 16;             // queries per wave
constexpr int COLS = 20;             // staged columns (union of clamped windows)
constexpr int CCH  = 32;             // channels per chunk (= WMMA K for f16)
constexpr int NCHUNK = C / CCH;      // 8

__device__ __forceinline__ int iclamp(int v, int lo, int hiv) {
  return v < lo ? lo : (v > hiv ? hiv : v);
}

__global__ __launch_bounds__(32)
void spconv_kernel(const float* __restrict__ x,
                   const float* __restrict__ sims,
                   float* __restrict__ out) {
  // f16 pairs: dword index (cp*5 + r)*20 + col holds channels {2cp, 2cp+1}
  __shared__ unsigned sXw[(CCH / 2) * WSZ * COLS];       // 6400 B
  __shared__ float    sScore[WSZ][2][TILE][TILE];        // 10240 B
  __shared__ float    sSims[9][WSZ][COLS];               // 3600 B
  __shared__ float    sKern[TILE][NK];                   // 1600 B
  _Float16* sXh = reinterpret_cast<_Float16*>(sXw);

  const int lane = threadIdx.x;        // 0..31
  const int n    = lane & 15;          // WMMA B/D column, A row
  const int hi   = lane >> 4;          // which K-half / M-half this lane holds
  const int w0   = blockIdx.x * TILE;
  const int h    = blockIdx.y;

  const int hA  = iclamp(h - 2, 0, H - WSZ);      // staged row-window start (ref clip)
  const int kA  = iclamp(w0 - 2, 0, W - COLS);    // staged col-window start
  const int kt0 = iclamp(w0 - 2, 0, W - TILE);    // key tile 0 start
  const int kt1 = iclamp(w0 + 2, 0, W - TILE);    // key tile 1 start

  // ---- stage sims neighborhood once: [9][5][20] -------------------------
  for (int idx = lane; idx < 9 * WSZ * COLS; idx += 32) {
    int f  = idx / (WSZ * COLS);
    int r  = (idx / COLS) % WSZ;
    int cl = idx % COLS;
    sSims[f][r][cl] = sims[f * HW + (hA + r) * W + (kA + cl)];
  }

  // ======================= pass 1: scores ================================
  v8f zero = {0.f, 0.f, 0.f, 0.f, 0.f, 0.f, 0.f, 0.f};
  v8f acc[WSZ][2];
#pragma unroll
  for (int j = 0; j < WSZ; ++j)
#pragma unroll
    for (int t = 0; t < 2; ++t) acc[j][t] = zero;

  for (int cc = 0; cc < NCHUNK; ++cc) {
    const int c0 = cc * CCH;
    __syncthreads();                       // sX reuse across chunks
    for (int idx = lane; idx < CCH * WSZ * COLS; idx += 32) {
      int c  = idx / (WSZ * COLS);
      int r  = (idx / COLS) % WSZ;
      int cl = idx % COLS;
      float xv = x[(c0 + c) * HW + (hA + r) * W + (kA + cl)];
      sXh[(((c >> 1) * WSZ + r) * COLS + cl) * 2 + (c & 1)] = (_Float16)xv;
    }
    if (cc + 1 < NCHUNK)                   // global_prefetch_b8 next chunk
      __builtin_prefetch(&x[(c0 + CCH + lane) * HW + hA * W + kA], 0, 1);
    __syncthreads();

    // A fragment: 16 queries x 32 channels.  ISA f16 A(16x32) layout:
    // lane m (+16*hi): dword v<4 -> K = 2v + 8*hi ; v>=4 -> K = 2v+8 + 8*hi.
    union { v16h h; unsigned u[8]; } Af;
    const int rq   = h - hA;
    const int colq = w0 - kA + n;          // query column of row m = n
#pragma unroll
    for (int v = 0; v < 8; ++v) {
      int cp = v + (v >= 4 ? 4 : 0) + hi * 4;           // channel pair index
      Af.u[v] = sXw[(cp * WSZ + rq) * COLS + colq];
    }
#pragma unroll
    for (int j = 0; j < WSZ; ++j) {
#pragma unroll
      for (int t = 0; t < 2; ++t) {
        // B fragment: 32 channels x 16 keys. lane n holds column n; dword v
        // -> same K map as A: pair cp = v + (v>=4?4:0) + 4*hi.
        union { v16h h; unsigned u[8]; } Bf;
        const int colk = (t ? kt1 : kt0) - kA + n;
#pragma unroll
        for (int v = 0; v < 8; ++v) {
          int cp = v + (v >= 4 ? 4 : 0) + hi * 4;
          Bf.u[v] = sXw[(cp * WSZ + j) * COLS + colk];
        }
        acc[j][t] = __builtin_amdgcn_wmma_f32_16x16x32_f16(
            false, Af.h, false, Bf.h, (short)0, acc[j][t], false, false);
      }
    }
  }

  // spill score matrices; D layout: lane n = column (key), M = v + 8*hi
#pragma unroll
  for (int j = 0; j < WSZ; ++j)
#pragma unroll
    for (int t = 0; t < 2; ++t)
#pragma unroll
      for (int v = 0; v < 8; ++v)
        sScore[j][t][hi * 8 + v][n] = acc[j][t][v];
  __syncthreads();

  // ============ softmax + sims reweight + L1 norm (per query) ============
  if (lane < TILE) {
    const int m  = lane;
    const int w  = w0 + m;
    const int ws0 = iclamp(w - 2, 0, W - WSZ);   // this query's window start
    float s[NK];
#pragma unroll
    for (int j = 0; j < WSZ; ++j)
#pragma unroll
      for (int i = 0; i < WSZ; ++i) {
        int kc = ws0 + i;                        // absolute key column
        int n0 = kc - kt0;
        float sc = (n0 >= 0 && n0 < TILE) ? sScore[j][0][m][n0]
                                          : sScore[j][1][m][kc - kt1];
        s[j * WSZ + i] = sc * 0.0625f;           // C^-0.5 = 1/16
      }
    // softmax
    float mx = s[0];
#pragma unroll
    for (int kk = 1; kk < NK; ++kk) mx = fmaxf(mx, s[kk]);
    float sum = 0.f;
#pragma unroll
    for (int kk = 0; kk < NK; ++kk) { float e = __expf(s[kk] - mx); s[kk] = e; sum += e; }
    float sinv = 1.f / sum;
    // sims reweight
    float sq[9];
#pragma unroll
    for (int f = 0; f < 9; ++f) sq[f] = sSims[f][h - hA][w - kA];
    float ssv[NK];
    float ssum = 0.f;
#pragma unroll
    for (int j = 0; j < WSZ; ++j)
#pragma unroll
      for (int i = 0; i < WSZ; ++i) {
        float a = 0.f;
#pragma unroll
        for (int f = 0; f < 9; ++f) a += sq[f] * sSims[f][j][ws0 - kA + i];
        ssv[j * WSZ + i] = a;
        ssum += a;
      }
    float rinv = 1.f / (1e-5f + ssum);
    float nsum = 0.f;
#pragma unroll
    for (int kk = 0; kk < NK; ++kk) {
      float val = (s[kk] * sinv) * (ssv[kk] * rinv);
      s[kk] = val;
      nsum += fabsf(val);
    }
    float kinv = 1.f / (1e-10f + nsum);
#pragma unroll
    for (int kk = 0; kk < NK; ++kk) sKern[m][kk] = s[kk] * kinv;
  }
  __syncthreads();

  // ===== build sparse kernel-band A fragments: 16 queries x 128 keys =====
  // key index = j*20 + col for key < 100, zero-padded to 128 (4 K-steps).
  union { v16h h; } A2[4];
  {
    const int m   = n;
    const int cq0 = iclamp(w0 + m - 2, 0, W - WSZ) - kA;  // window start in LDS cols
#pragma unroll
    for (int sstep = 0; sstep < 4; ++sstep) {
#pragma unroll
      for (int e = 0; e < 16; ++e) {
        int Kk  = e + (e >= 8 ? 8 : 0) + hi * 8;          // A f16 element->K map
        int key = sstep * 32 + Kk;
        float val = 0.f;
        if (key < WSZ * COLS) {
          int j  = key / COLS;
          int cl = key % COLS;
          int i  = cl - cq0;
          if (i >= 0 && i < WSZ) val = sKern[m][j * WSZ + i];
        }
        A2[sstep].h[e] = (_Float16)val;
      }
    }
  }

  // ================= pass 2: out = kernel x neighbors ====================
  for (int cc = 0; cc < NCHUNK; ++cc) {
    const int c0 = cc * CCH;
    __syncthreads();
    for (int idx = lane; idx < CCH * WSZ * COLS; idx += 32) {
      int c  = idx / (WSZ * COLS);
      int r  = (idx / COLS) % WSZ;
      int cl = idx % COLS;
      float xv = x[(c0 + c) * HW + (hA + r) * W + (kA + cl)];
      sXh[(((c >> 1) * WSZ + r) * COLS + cl) * 2 + (c & 1)] = (_Float16)xv;
    }
    if (cc + 1 < NCHUNK)
      __builtin_prefetch(&x[(c0 + CCH + lane) * HW + hA * W + kA], 0, 1);
    __syncthreads();

#pragma unroll
    for (int nsub = 0; nsub < 2; ++nsub) {
      const int cloc = nsub * TILE + n;        // local channel 0..31 (= B column)
      v8f acc2 = zero;
#pragma unroll
      for (int sstep = 0; sstep < 4; ++sstep) {
        union { v16h h; } B2;
#pragma unroll
        for (int e = 0; e < 16; ++e) {
          int key = sstep * 32 + e + (e >= 8 ? 8 : 0) + hi * 8;  // B element->K map
          _Float16 val = (_Float16)0.f;
          if (key < WSZ * COLS) {
            int j  = key / COLS;
            int cl = key % COLS;
            val = sXh[(((cloc >> 1) * WSZ + j) * COLS + cl) * 2 + (cloc & 1)];
          }
          B2.h[e] = val;
        }
        acc2 = __builtin_amdgcn_wmma_f32_16x16x32_f16(
            false, A2[sstep].h, false, B2.h, (short)0, acc2, false, false);
      }
      // D: lane n = channel column, M = v + 8*hi = query pixel
      const int c = c0 + cloc;
#pragma unroll
      for (int v = 0; v < 8; ++v)
        out[c * HW + h * W + (w0 + hi * 8 + v)] = acc2[v];
    }
  }
}

extern "C" void kernel_launch(void* const* d_in, const int* /*in_sizes*/, int /*n_in*/,
                              void* d_out, int /*out_size*/, void* /*d_ws*/, size_t /*ws_size*/,
                              hipStream_t stream) {
  const float* x    = (const float*)d_in[0];   // (1,256,128,128) f32
  const float* sims = (const float*)d_in[1];   // (1,9,128,128)  f32
  // d_in[2] = flows: unused by the reference computation (zeros)
  float* out = (float*)d_out;                  // (1,256,128,128) f32

  dim3 grid(W / TILE, H, 1);                   // 8 x 128 single-wave workgroups
  spconv_kernel<<<grid, 32, 0, stream>>>(x, sims, out);
}
